// MoEWrapper_22995254902973
// MI455X (gfx1250) — compile-verified, matrix operand order checked by reading, not run on hardware
//
#include <hip/hip_runtime.h>
#include <hip/hip_bf16.h>
#include <stdint.h>

// ---------------- problem constants ----------------
constexpr int Bv = 4, Sv = 512, Hc = 1024;
constexpr int Ec = 8, Ic = 2048;
constexpr int Tc = Bv * Sv;              // 2048 tokens
constexpr int KTH = Hc / 32;             // 32  k-tiles over H
constexpr int KTI = Ic / 32;             // 64  k-tiles over I

typedef __attribute__((ext_vector_type(16))) __bf16 v16bf;
typedef __attribute__((ext_vector_type(8)))  float  v8f;
typedef __attribute__((ext_vector_type(4)))  int    v4i_;

// ---------------- async global->LDS copy (CDNA5) ----------------
#if defined(__has_builtin)
#if __has_builtin(__builtin_amdgcn_global_load_async_to_lds_b128)
#define MOE_ASYNC_LDS 1
#endif
#endif
#ifndef MOE_ASYNC_LDS
#define MOE_ASYNC_LDS 0
#endif

static __device__ __forceinline__ void cp16(unsigned short* l, const unsigned short* g) {
#if MOE_ASYNC_LDS
  __builtin_amdgcn_global_load_async_to_lds_b128(
      (__attribute__((address_space(1))) v4i_*)(unsigned long long)(uintptr_t)g,
      (__attribute__((address_space(3))) v4i_*)(unsigned int)(uintptr_t)l,
      0, 0);
#else
  *reinterpret_cast<uint4*>(l) = *reinterpret_cast<const uint4*>(g);
#endif
}

static __device__ __forceinline__ void wait_async0() {
#if MOE_ASYNC_LDS
#if __has_builtin(__builtin_amdgcn_s_wait_asynccnt)
  __builtin_amdgcn_s_wait_asynccnt(0);
#else
  asm volatile("s_wait_asynccnt 0x0" ::: "memory");
#endif
#endif
}

// fp32 -> bf16 round-to-nearest-even (raw u16)
static __device__ __forceinline__ unsigned short f32_to_bf16(float f) {
  unsigned int u = __float_as_uint(f);
  u += 0x7FFFu + ((u >> 16) & 1u);
  return (unsigned short)(u >> 16);
}

// 16x32 bf16 fragment from LDS tile stored [row][32]:
// lane L<16: row, K = 0..7 & 16..23 ; lane L>=16: K = 8..15 & 24..31
static __device__ __forceinline__ v16bf load_frag16x32(const unsigned short* base,
                                                       int row, int kb) {
  v16bf f;
  uint4* p = reinterpret_cast<uint4*>(&f);
  const unsigned short* r0 = base + row * 32;
  p[0] = *reinterpret_cast<const uint4*>(r0 + kb);
  p[1] = *reinterpret_cast<const uint4*>(r0 + kb + 16);
  return f;
}

static __device__ __forceinline__ v8f wmma_bf16(v16bf a, v16bf b, v8f c) {
  return __builtin_amdgcn_wmma_f32_16x16x32_bf16(false, a, false, b,
                                                 (short)0, c, false, false);
}

// ---------------- kernel: zero output ----------------
__global__ void moe_zero(float4* __restrict__ p, int n4) {
  int i = blockIdx.x * blockDim.x + threadIdx.x;
  if (i < n4) p[i] = make_float4(0.f, 0.f, 0.f, 0.f);
}

// ---------------- kernel: pack fp32 [e][K][N] -> bf16 LDS-tile order ----------------
// tile (kt, nt) of 32(K) x 128(N), stored n-major [128][32], tiles ordered (nt*KT + kt)
__global__ __launch_bounds__(256) void moe_pack(
    const float* __restrict__ src, unsigned short* __restrict__ dst,
    int Kdim, int Ndim, int nExp) {
  const unsigned int total = (unsigned int)nExp * (unsigned int)(Kdim >> 2) * (unsigned int)Ndim;
  const unsigned int idx = blockIdx.x * 256u + threadIdx.x;
  if (idx >= total) return;
  const int n  = (int)(idx % (unsigned int)Ndim);
  const unsigned int t2 = idx / (unsigned int)Ndim;
  const int k4 = (int)(t2 % (unsigned int)(Kdim >> 2));
  const int e  = (int)(t2 / (unsigned int)(Kdim >> 2));
  const int k  = k4 << 2;
  const size_t mat = (size_t)Kdim * (size_t)Ndim;
  const float* s = src + (size_t)e * mat + (size_t)k * Ndim + n;
  const int KT = Kdim >> 5;
  const int kt = k >> 5, kl = k & 31, nt = n >> 7, nl = n & 127;
  unsigned short* d = dst + (size_t)e * mat +
                      ((size_t)(nt * KT + kt)) * 4096 + nl * 32 + kl;
  const unsigned int b0 = f32_to_bf16(s[0]);
  const unsigned int b1 = f32_to_bf16(s[(size_t)Ndim]);
  const unsigned int b2 = f32_to_bf16(s[(size_t)2 * Ndim]);
  const unsigned int b3 = f32_to_bf16(s[(size_t)3 * Ndim]);
  uint2 o;
  o.x = b0 | (b1 << 16);
  o.y = b2 | (b3 << 16);
  *reinterpret_cast<uint2*>(d) = o;
}

// ---------------- kernel: router + x->bf16 (packed tiles) ----------------
__global__ __launch_bounds__(256) void moe_router(
    const float* __restrict__ x,        // [T,H]
    const float* __restrict__ rw,       // [E,H]
    float* __restrict__ aff,            // [T,E]
    unsigned short* __restrict__ xb) {  // packed [T/64][H/32][64][32] bf16
  const int wave = threadIdx.x >> 5;
  const int lane = threadIdx.x & 31;
  const int t = blockIdx.x * 8 + wave;
  const float* xr = x + (size_t)t * Hc;

  for (int j = lane; j < Hc; j += 32)
    xb[((size_t)(t >> 6) * KTH + (j >> 5)) * 2048 + (t & 63) * 32 + (j & 31)] =
        f32_to_bf16(xr[j]);

  float acc[Ec];
#pragma unroll
  for (int e = 0; e < Ec; ++e) acc[e] = 0.f;
  for (int j = lane; j < Hc; j += 32) {
    const float xv = xr[j];
#pragma unroll
    for (int e = 0; e < Ec; ++e) acc[e] += xv * rw[e * Hc + j];
  }
#pragma unroll
  for (int e = 0; e < Ec; ++e)
    for (int off = 16; off > 0; off >>= 1)
      acc[e] += __shfl_xor(acc[e], off, 32);

  if (lane == 0) {
    float m = acc[0];
#pragma unroll
    for (int e = 1; e < Ec; ++e) m = fmaxf(m, acc[e]);
    float p[Ec], s = 0.f;
#pragma unroll
    for (int e = 0; e < Ec; ++e) { p[e] = __expf(acc[e] - m); s += p[e]; }
    const float inv = 1.f / s;
#pragma unroll
    for (int e = 0; e < Ec; ++e) p[e] *= inv;
    int i1 = 0;
#pragma unroll
    for (int e = 1; e < Ec; ++e) if (p[e] > p[i1]) i1 = e;
    int i2 = (i1 == 0) ? 1 : 0;
#pragma unroll
    for (int e = 0; e < Ec; ++e) if (e != i1 && p[e] > p[i2]) i2 = e;
    const float norm = 1.f / (p[i1] + p[i2]);
#pragma unroll
    for (int e = 0; e < Ec; ++e)
      aff[(size_t)t * Ec + e] = (e == i1) ? p[i1] * norm
                              : (e == i2) ? p[i2] * norm : 0.f;
  }
}

// ---------------- kernel: GEMM1 (gate/up + swiglu), double-buffered ----------------
__global__ __launch_bounds__(256) void moe_gemm1(
    const unsigned short* __restrict__ xb,  // packed x tiles
    const unsigned short* __restrict__ pg,  // packed gate [I/128][H/32][128][32]
    const unsigned short* __restrict__ pu,  // packed up
    unsigned short* __restrict__ act) {     // packed [T/64][I/32][64][32]
  __shared__ unsigned short smem[2][10240];  // X:0..2047  Wg:2048..6143  Wu:6144..10239

  const int tid = threadIdx.x;
  const int bx = blockIdx.x;   // I tile (128 wide)
  const int by = blockIdx.y;   // token tile (64)
  const int wave = tid >> 5, lane = tid & 31;
  const int wm = wave & 1, wn = wave >> 1;
  const int lrow = lane & 15;
  const int kb = (lane >> 4) << 3;

  v8f accg[2][2] = {};
  v8f accu[2][2] = {};

  auto stage = [&](int b, int kt) {
    const unsigned short* xs = xb + ((size_t)by * KTH + kt) * 2048;
    const unsigned short* gs = pg + ((size_t)bx * KTH + kt) * 4096;
    const unsigned short* us = pu + ((size_t)bx * KTH + kt) * 4096;
    unsigned short* l = smem[b];
    const int o = tid * 8;
    cp16(l + o,         xs + o);
    cp16(l + 2048 + o,  gs + o);
    cp16(l + 4096 + o,  gs + 2048 + o);
    cp16(l + 6144 + o,  us + o);
    cp16(l + 8192 + o,  us + 2048 + o);
  };

  stage(0, 0);
  for (int kt = 0; kt < KTH; ++kt) {
    wait_async0();
    __syncthreads();
    if (kt + 1 < KTH) stage((kt + 1) & 1, kt + 1);
    const unsigned short* l = smem[kt & 1];
    // hoisted B fragments: load each once per K-step
    const int nr0 = wn * 32 + lrow;
    const v16bf bg0 = load_frag16x32(l + 2048, nr0,      kb);
    const v16bf bg1 = load_frag16x32(l + 2048, nr0 + 16, kb);
    const v16bf bu0 = load_frag16x32(l + 6144, nr0,      kb);
    const v16bf bu1 = load_frag16x32(l + 6144, nr0 + 16, kb);
#pragma unroll
    for (int si = 0; si < 2; ++si) {
      const v16bf a = load_frag16x32(l, wm * 32 + si * 16 + lrow, kb);
      accg[si][0] = wmma_bf16(a, bg0, accg[si][0]);
      accg[si][1] = wmma_bf16(a, bg1, accg[si][1]);
      accu[si][0] = wmma_bf16(a, bu0, accu[si][0]);
      accu[si][1] = wmma_bf16(a, bu1, accu[si][1]);
    }
  }

  // epilogue: act = silu(gate) * up -> packed bf16
#pragma unroll
  for (int si = 0; si < 2; ++si)
#pragma unroll
    for (int sn = 0; sn < 2; ++sn) {
      const int col = bx * 128 + wn * 32 + sn * 16 + lrow;
#pragma unroll
      for (int r = 0; r < 8; ++r) {
        const float g = accg[si][sn][r];
        const float u = accu[si][sn][r];
        const float sv = g / (1.f + __expf(-g));
        const int trow = wm * 32 + si * 16 + ((lane < 16) ? r : r + 8);  // local 0..63
        act[((size_t)by * KTI + (col >> 5)) * 2048 + trow * 32 + (col & 31)] =
            f32_to_bf16(sv * u);
      }
    }
}

// ---------------- kernel: GEMM2 (down proj, scaled accumulate), double-buffered ----
__global__ __launch_bounds__(256) void moe_gemm2(
    const unsigned short* __restrict__ act,  // packed act tiles
    const unsigned short* __restrict__ pd,   // packed down [H/128][I/32][128][32]
    const float* __restrict__ aff,           // [T,E]
    float* __restrict__ out,                 // [T,H]
    int e) {
  __shared__ unsigned short smem[2][6144];   // A:0..2047  Wd:2048..6143

  const int tid = threadIdx.x;
  const int bx = blockIdx.x;   // H tile (128 wide)
  const int by = blockIdx.y;   // token tile (64)
  const int wave = tid >> 5, lane = tid & 31;
  const int wm = wave & 1, wn = wave >> 1;
  const int lrow = lane & 15;
  const int kb = (lane >> 4) << 3;

  v8f acc[2][2] = {};

  auto stage = [&](int b, int kt) {
    const unsigned short* as = act + ((size_t)by * KTI + kt) * 2048;
    const unsigned short* ds = pd + ((size_t)bx * KTI + kt) * 4096;
    unsigned short* l = smem[b];
    const int o = tid * 8;
    cp16(l + o,         as + o);
    cp16(l + 2048 + o,  ds + o);
    cp16(l + 4096 + o,  ds + 2048 + o);
  };

  stage(0, 0);
  for (int kt = 0; kt < KTI; ++kt) {
    wait_async0();
    __syncthreads();
    if (kt + 1 < KTI) stage((kt + 1) & 1, kt + 1);
    const unsigned short* l = smem[kt & 1];
    const int nr0 = wn * 32 + lrow;
    const v16bf b0 = load_frag16x32(l + 2048, nr0,      kb);
    const v16bf b1 = load_frag16x32(l + 2048, nr0 + 16, kb);
#pragma unroll
    for (int si = 0; si < 2; ++si) {
      const v16bf a = load_frag16x32(l, wm * 32 + si * 16 + lrow, kb);
      acc[si][0] = wmma_bf16(a, b0, acc[si][0]);
      acc[si][1] = wmma_bf16(a, b1, acc[si][1]);
    }
  }

#pragma unroll
  for (int si = 0; si < 2; ++si)
#pragma unroll
    for (int sn = 0; sn < 2; ++sn) {
      const int col = bx * 128 + wn * 32 + sn * 16 + lrow;
#pragma unroll
      for (int r = 0; r < 8; ++r) {
        const int trow = by * 64 + wm * 32 + si * 16 + ((lane < 16) ? r : r + 8);
        const float s = aff[(size_t)trow * Ec + e];
        float* po = out + (size_t)trow * Hc + col;
        *po += s * acc[si][sn][r];
      }
    }
}

// ---------------- host launcher ----------------
extern "C" void kernel_launch(void* const* d_in, const int* in_sizes, int n_in,
                              void* d_out, int out_size, void* d_ws, size_t ws_size,
                              hipStream_t stream) {
  (void)in_sizes; (void)n_in; (void)out_size;
  const float* x  = (const float*)d_in[0];   // [B,S,H]
  const float* rw = (const float*)d_in[1];   // [E,H]
  const float* wg = (const float*)d_in[2];   // [E,H,I]
  const float* wu = (const float*)d_in[3];   // [E,H,I]
  const float* wd = (const float*)d_in[4];   // [E,I,H]
  float* out = (float*)d_out;

  const size_t matHI = (size_t)Hc * Ic;      // 2,097,152 elements
  char* ws = (char*)d_ws;
  size_t off = 0;
  unsigned short* xb  = (unsigned short*)(ws + off); off += (size_t)Tc * Hc * 2;  // 4 MB
  float*          aff = (float*)(ws + off);          off += (size_t)Tc * Ec * 4;  // 64 KB
  unsigned short* ab  = (unsigned short*)(ws + off); off += (size_t)Tc * Ic * 2;  // 8 MB
  unsigned short* wsl = (unsigned short*)(ws + off);                               // weights

  const size_t need_full = off + 3 * (size_t)Ec * matHI * 2;   // ~113.3 MB
  const bool full = ws_size >= need_full;

  const int packBlocksHI1 = (int)(((size_t)1 * (Hc / 4) * Ic + 255) / 256);
  const int packBlocksHI8 = (int)(((size_t)Ec * (Hc / 4) * Ic + 255) / 256);
  const int packBlocksIH1 = (int)(((size_t)1 * (Ic / 4) * Hc + 255) / 256);
  const int packBlocksIH8 = (int)(((size_t)Ec * (Ic / 4) * Hc + 255) / 256);

  unsigned short* PG = wsl;
  unsigned short* PU = full ? (wsl + Ec * matHI) : (wsl + matHI);
  unsigned short* PD = full ? (wsl + 2 * Ec * matHI) : (wsl + 2 * matHI);

  if (full) {
    moe_pack<<<packBlocksHI8, 256, 0, stream>>>(wg, PG, Hc, Ic, Ec);
    moe_pack<<<packBlocksHI8, 256, 0, stream>>>(wu, PU, Hc, Ic, Ec);
    moe_pack<<<packBlocksIH8, 256, 0, stream>>>(wd, PD, Ic, Hc, Ec);
  }

  const int n4 = (Tc * Hc) / 4;
  moe_zero<<<(n4 + 255) / 256, 256, 0, stream>>>((float4*)out, n4);
  moe_router<<<Tc / 8, 256, 0, stream>>>(x, rw, aff, xb);

  for (int e = 0; e < Ec; ++e) {
    const unsigned short *pge, *pue, *pde;
    if (full) {
      pge = PG + (size_t)e * matHI;
      pue = PU + (size_t)e * matHI;
      pde = PD + (size_t)e * matHI;
    } else {
      moe_pack<<<packBlocksHI1, 256, 0, stream>>>(wg + (size_t)e * matHI, PG, Hc, Ic, 1);
      moe_pack<<<packBlocksHI1, 256, 0, stream>>>(wu + (size_t)e * matHI, PU, Hc, Ic, 1);
      moe_pack<<<packBlocksIH1, 256, 0, stream>>>(wd + (size_t)e * matHI, PD, Ic, Hc, 1);
      pge = PG; pue = PU; pde = PD;
    }
    moe_gemm1<<<dim3(Ic / 128, Tc / 64), 256, 0, stream>>>(xb, pge, pue, ab);
    moe_gemm2<<<dim3(Hc / 128, Tc / 64), 256, 0, stream>>>(ab, pde, aff, out, e);
  }
}
